// LPFormer_59957743452825
// MI455X (gfx1250) — matrix operand order repaired and checked by reading, not compiled, not run on hardware
//
#include <hip/hip_runtime.h>
#include <hip/hip_bf16.h>

#define HID    128
#define OUTC   128
#define NLINK  16384
#define NEDGE  1048576
#define NNODE  16384
#define NEG_SLOPE 0.2f

typedef _Float16 v16h __attribute__((ext_vector_type(16)));
typedef _Float16 v8h  __attribute__((ext_vector_type(8)));
typedef float    v8f  __attribute__((ext_vector_type(8)));
typedef float    v4f  __attribute__((ext_vector_type(4)));

// order-preserving float<->uint mapping for atomicMax on floats
__device__ __forceinline__ unsigned fenc(float f) {
    unsigned b = __float_as_uint(f);
    return (b & 0x80000000u) ? ~b : (b | 0x80000000u);
}
__device__ __forceinline__ float fdec(unsigned u) {
    unsigned b = (u & 0x80000000u) ? (u & 0x7FFFFFFFu) : ~u;
    return __uint_as_float(b);
}
__device__ __forceinline__ _Float16 u16_as_half(unsigned short u) {
    return __builtin_bit_cast(_Float16, u);
}

// Async-copy a linear 32 KB block (global f16 image -> LDS), 256 threads.
// Each lane moves 16 B per instruction; tracked by ASYNCcnt. After
// s_wait_asynccnt 0 each wave's copies have landed; the caller's
// __syncthreads() makes the whole block's LDS image coherent.
__device__ __forceinline__ void stage32k_async(_Float16* lds, const _Float16* gsrc, int tid) {
    unsigned lbase = (unsigned)(uintptr_t)lds;   // low 32 bits == DS offset
    const char* g = (const char*)gsrc;
    #pragma unroll
    for (int it = 0; it < 8; ++it) {
        unsigned off = (unsigned)(tid * 16 + it * 4096);
        asm volatile("global_load_async_to_lds_b128 %0, %1, off"
                     :: "v"(lbase + off), "v"((const void*)(g + off))
                     : "memory");
    }
    asm volatile("s_wait_asynccnt 0x0" ::: "memory");
}

// ---------------------------------------------------------------- init ----
__global__ __launch_bounds__(256) void lpf_init(float* __restrict__ outacc,
                                                float* __restrict__ denom,
                                                unsigned* __restrict__ mkey) {
    int i = blockIdx.x * 256 + threadIdx.x;
    if (i < NLINK * OUTC) outacc[i] = 0.0f;
    if (i < NLINK) { denom[i] = 0.0f; mkey[i] = 0u; }   // 0 == below all encoded floats
}

// -------------------------- one-time weight convert/transpose to f16 -----
// Wlt_h[n*128 + k]              = W_l[k][n]                 (32 KB)
// Wrt_h[p*16384 + n*128 + kl]   = W_r[p*128 + kl][n]        (2 x 32 KB)
__global__ __launch_bounds__(256) void lpf_wconv(const float* __restrict__ W_l,
                                                 const float* __restrict__ W_r,
                                                 _Float16* __restrict__ Wlt_h,
                                                 _Float16* __restrict__ Wrt_h) {
    int i = blockIdx.x * 256 + threadIdx.x;
    if (i < 128 * 128) {
        int k = i >> 7, n = i & 127;
        Wlt_h[n * 128 + k] = (_Float16)W_l[i];
    }
    if (i < 256 * 128) {
        int k = i >> 7, n = i & 127;
        int p = k >> 7, kl = k & 127;
        Wrt_h[p * 16384 + n * 128 + kl] = (_Float16)W_r[i];
    }
}

// ------------------------------------------------- per-link e = (e1+e2)W_l
// one wave computes a 16-link x 128-col tile, K=128 via 4 WMMA k-steps
__global__ __launch_bounds__(256) void lpf_elink(const float* __restrict__ edge_feats,
                                                 const _Float16* __restrict__ Wlt_h,
                                                 const float* __restrict__ b_l,
                                                 float* __restrict__ e_link) {
    __shared__ _Float16 Wlt[128 * 128];            // Wlt[n*128 + k], 32 KB
    int tid = threadIdx.x;
    stage32k_async(Wlt, Wlt_h, tid);
    __syncthreads();

    int wave = tid >> 5, lane = tid & 31;
    int hi = lane >> 4, ln = lane & 15;
    int row0 = (blockIdx.x * 8 + wave) * 16;
    const float* arow = edge_feats + (size_t)(row0 + ln) * (2 * HID);

    v8f acc[8];
    #pragma unroll
    for (int t = 0; t < 8; ++t)
        #pragma unroll
        for (int i = 0; i < 8; ++i) acc[t][i] = 0.0f;

    #pragma unroll
    for (int ks = 0; ks < 4; ++ks) {
        v16h a;
        #pragma unroll
        for (int c = 0; c < 2; ++c) {
            int k = ks * 32 + c * 16 + hi * 8;     // absolute K for this 8-chunk
            v4f p0 = *(const v4f*)(arow + k);
            v4f p1 = *(const v4f*)(arow + k + 4);
            v4f q0 = *(const v4f*)(arow + HID + k);
            v4f q1 = *(const v4f*)(arow + HID + k + 4);
            #pragma unroll
            for (int j = 0; j < 4; ++j) {
                a[c * 8 + j]     = (_Float16)(p0[j] + q0[j]);
                a[c * 8 + 4 + j] = (_Float16)(p1[j] + q1[j]);
            }
        }
        #pragma unroll
        for (int t = 0; t < 8; ++t) {
            int n = t * 16 + ln;
            const v8h* bp = (const v8h*)(&Wlt[n * 128 + ks * 32 + hi * 8]);
            v8h lo = bp[0], hs = bp[2];
            v16h b;
            #pragma unroll
            for (int j = 0; j < 8; ++j) { b[j] = lo[j]; b[8 + j] = hs[j]; }
            acc[t] = __builtin_amdgcn_wmma_f32_16x16x32_f16(
                false, a, false, b, (short)0, acc[t], false, false);
        }
    }
    #pragma unroll
    for (int t = 0; t < 8; ++t) {
        int n = t * 16 + ln;
        float bb = 2.0f * b_l[n];
        #pragma unroll
        for (int r = 0; r < 8; ++r)
            e_link[(size_t)(row0 + hi * 8 + r) * OUTC + n] = acc[t][r] + bb;
    }
}

// ------------------------------------- main pass: x_j GEMM + alpha + max --
// one wave: 16 edges x 128 cols, K=256 (k 0..127 node gather, 128..255 ppr)
__global__ __launch_bounds__(256) void lpf_pass1(const int* __restrict__ idx,
                                                 const float* __restrict__ node_feats,
                                                 const float* __restrict__ ppr,
                                                 const _Float16* __restrict__ Wrt_h,
                                                 const float* __restrict__ b_r,
                                                 const float* __restrict__ e_link,
                                                 const float* __restrict__ att,
                                                 _Float16* __restrict__ xj,
                                                 float* __restrict__ alpha,
                                                 unsigned* __restrict__ mkey) {
    __shared__ _Float16 Wrt[128 * 128];            // one K-half of W_r^T, 32 KB
    int tid = threadIdx.x;
    int wave = tid >> 5, lane = tid & 31;
    int hi = lane >> 4, ln = lane & 15;
    int e0 = (blockIdx.x * 8 + wave) * 16;
    int myedge = e0 + ln;
    int seg_m = idx[myedge];
    int nbr_m = idx[NEDGE + myedge];
    const float* nrow = node_feats + (size_t)nbr_m * HID;
    const float* prow = ppr + (size_t)myedge * HID;

    v8f acc[8];
    #pragma unroll
    for (int t = 0; t < 8; ++t)
        #pragma unroll
        for (int i = 0; i < 8; ++i) acc[t][i] = 0.0f;

    #pragma unroll
    for (int phase = 0; phase < 2; ++phase) {
        if (phase) __syncthreads();                // all waves done reading phase 0
        stage32k_async(Wrt, Wrt_h + phase * 16384, tid);
        __syncthreads();
        const float* src_base = phase ? prow : nrow;
        #pragma unroll
        for (int ksl = 0; ksl < 4; ++ksl) {
            v16h a;
            #pragma unroll
            for (int c = 0; c < 2; ++c) {
                int k = ksl * 32 + c * 16 + hi * 8;   // K local to this 128-half
                v4f p0 = *(const v4f*)(src_base + k);
                v4f p1 = *(const v4f*)(src_base + k + 4);
                #pragma unroll
                for (int j = 0; j < 4; ++j) {
                    a[c * 8 + j]     = (_Float16)p0[j];
                    a[c * 8 + 4 + j] = (_Float16)p1[j];
                }
            }
            #pragma unroll
            for (int t = 0; t < 8; ++t) {
                int n = t * 16 + ln;
                const v8h* bp = (const v8h*)(&Wrt[n * 128 + ksl * 32 + hi * 8]);
                v8h lo = bp[0], hs = bp[2];
                v16h b;
                #pragma unroll
                for (int j = 0; j < 8; ++j) { b[j] = lo[j]; b[8 + j] = hs[j]; }
                acc[t] = __builtin_amdgcn_wmma_f32_16x16x32_f16(
                    false, a, false, b, (short)0, acc[t], false, false);
            }
        }
    }

    // broadcast segment ids of the 8 rows this lane's accumulators cover
    int segs[8];
    #pragma unroll
    for (int r = 0; r < 8; ++r) segs[r] = __shfl(seg_m, hi * 8 + r, 32);

    float apart[8];
    #pragma unroll
    for (int r = 0; r < 8; ++r) apart[r] = 0.0f;

    #pragma unroll
    for (int t = 0; t < 8; ++t) {
        int n = t * 16 + ln;
        float br = b_r[n];
        float at = att[n];
        #pragma unroll
        for (int r = 0; r < 8; ++r) {
            float v = acc[t][r] + br;                           // x_j element
            int row = e0 + hi * 8 + r;
            xj[(size_t)row * OUTC + n] = (_Float16)v;           // f16 spill
            float ev = e_link[(size_t)segs[r] * OUTC + n];      // L2-hot gather
            float x = v * ev;
            x = (x >= 0.0f) ? x : NEG_SLOPE * x;                // leaky relu
            apart[r] += x * at;
        }
    }
    // reduce alpha over the 16 lanes of each half-wave, then publish
    #pragma unroll
    for (int r = 0; r < 8; ++r) {
        float s = apart[r];
        s += __shfl_xor(s, 8, 16);
        s += __shfl_xor(s, 4, 16);
        s += __shfl_xor(s, 2, 16);
        s += __shfl_xor(s, 1, 16);
        if (ln == r) {
            alpha[e0 + hi * 8 + r] = s;
            atomicMax(&mkey[segs[r]], fenc(s));
        }
    }
}

// --------------------------- scatter: exp, denom, weighted accumulation --
__global__ __launch_bounds__(256) void lpf_pass2(const int* __restrict__ idx,
                                                 const unsigned* __restrict__ xj_u32,
                                                 const float* __restrict__ alpha,
                                                 const unsigned* __restrict__ mkey,
                                                 float* __restrict__ denom,
                                                 float* __restrict__ outacc) {
    int wave = threadIdx.x >> 5, lane = threadIdx.x & 31;
    int e = blockIdx.x * 8 + wave;
    int seg = idx[e];
    float m = fdec(mkey[seg]);
    float a = __expf(alpha[e] - m);
    if (lane == 0) atomicAdd(&denom[seg], a);
    const unsigned* xr = xj_u32 + (size_t)e * (OUTC / 2);   // packed 2 x f16
    float* orow = outacc + (size_t)seg * OUTC;
    #pragma unroll
    for (int q = 0; q < 2; ++q) {
        int u = lane + q * 32;
        unsigned pk = xr[u];
        float h0 = (float)u16_as_half((unsigned short)(pk & 0xFFFFu));
        float h1 = (float)u16_as_half((unsigned short)(pk >> 16));
        atomicAdd(&orow[2 * u],     h0 * a);
        atomicAdd(&orow[2 * u + 1], h1 * a);
    }
}

// ----------------------------------------- finalize: divide + LayerNorm --
__global__ __launch_bounds__(128) void lpf_pass3(const float* __restrict__ denom,
                                                 const float* __restrict__ bias,
                                                 const float* __restrict__ gamma,
                                                 const float* __restrict__ beta,
                                                 float* __restrict__ out) {
    int l = blockIdx.x, c = threadIdx.x;
    float v = out[(size_t)l * OUTC + c] / (denom[l] + 1e-16f) + bias[c];
    float s = v, q = v * v;
    #pragma unroll
    for (int off = 16; off >= 1; off >>= 1) {
        s += __shfl_xor(s, off, 32);
        q += __shfl_xor(q, off, 32);
    }
    __shared__ float ss[4], sq[4];
    if ((c & 31) == 0) { ss[c >> 5] = s; sq[c >> 5] = q; }
    __syncthreads();
    float S = ss[0] + ss[1] + ss[2] + ss[3];
    float Q = sq[0] + sq[1] + sq[2] + sq[3];
    float mu  = S * (1.0f / OUTC);
    float var = Q * (1.0f / OUTC) - mu * mu;
    out[(size_t)l * OUTC + c] = (v - mu) * rsqrtf(var + 1e-5f) * gamma[c] + beta[c];
}

// -------------------------------------------------------------------------
extern "C" void kernel_launch(void* const* d_in, const int* in_sizes, int n_in,
                              void* d_out, int out_size, void* d_ws, size_t ws_size,
                              hipStream_t stream) {
    const int*   idx        = (const int*)d_in[0];     // [2, E]: seg then nbr
    const float* edge_feats = (const float*)d_in[1];   // [L, 256]
    const float* node_feats = (const float*)d_in[2];   // [N, 128]
    const float* ppr        = (const float*)d_in[3];   // [E, 128]
    const float* W_l        = (const float*)d_in[4];   // [128, 128]
    const float* b_l        = (const float*)d_in[5];   // [128]
    const float* W_r        = (const float*)d_in[6];   // [256, 128]
    const float* b_r        = (const float*)d_in[7];   // [128]
    const float* att        = (const float*)d_in[8];   // [128]
    const float* bias       = (const float*)d_in[9];   // [128]
    const float* ln_gamma   = (const float*)d_in[10];  // [128]
    const float* ln_beta    = (const float*)d_in[11];  // [128]
    float* out = (float*)d_out;                        // [L, 128]

    char* ws = (char*)d_ws;
    float*    e_link = (float*)(ws);                                       // 8 MB
    float*    alpha  = (float*)(ws + (size_t)8 * 1024 * 1024);             // 4 MB
    unsigned* mkey   = (unsigned*)(ws + (size_t)12 * 1024 * 1024);         // 64 KB
    float*    denom  = (float*)(ws + (size_t)12 * 1024 * 1024 + 65536);    // 64 KB
    _Float16* Wlt_h  = (_Float16*)(ws + (size_t)12 * 1024 * 1024 + 131072);// 32 KB
    _Float16* Wrt_h  = (_Float16*)(ws + (size_t)12 * 1024 * 1024 + 163840);// 64 KB
    _Float16* xj     = (_Float16*)(ws + (size_t)13 * 1024 * 1024);         // 256 MB

    lpf_init <<<(NLINK * OUTC) / 256, 256, 0, stream>>>(out, denom, mkey);
    lpf_wconv<<<128, 256, 0, stream>>>(W_l, W_r, Wlt_h, Wrt_h);
    lpf_elink<<<NLINK / 128, 256, 0, stream>>>(edge_feats, Wlt_h, b_l, e_link);
    lpf_pass1<<<NEDGE / 128, 256, 0, stream>>>(idx, node_feats, ppr, Wrt_h, b_r,
                                               e_link, att, xj, alpha, mkey);
    lpf_pass2<<<NEDGE / 8, 256, 0, stream>>>(idx, (const unsigned*)xj, alpha, mkey,
                                             denom, out);
    lpf_pass3<<<NLINK, 128, 0, stream>>>(denom, bias, ln_gamma, ln_beta, out);
}
